// DAttention_59914793779458
// MI455X (gfx1250) — matrix operand order repaired and checked by reading, not compiled, not run on hardware
//
#include <hip/hip_runtime.h>
#include <cfloat>

// ---------------------------------------------------------------------------
// Single-query gated cross-attention, algebraically refactored:
//   s[h,j]  = r_h . c_j + bias          (r_h = Wk_h^T q_h * SCALE)
//   Y_h     = sum_j softmax(s)[h,j] c_j
//   out     = Wo ( sigmoid(Wg qx + bg) * (Wv_h Y_h / l_h) ) + bo
// K/V are never materialized -> two streaming passes over context (HBM bound),
// score & weighted-sum GEMMs done with v_wmma_f32_16x16x32_bf16.
// ---------------------------------------------------------------------------

#define BATCH   16
#define NCTX    4096
#define DIM     512
#define HEADS   8
#define JBLK    256           // context rows per workgroup
#define NJB     (NCTX / JBLK) // 16 j-blocks per batch
#define CSTR    48            // padded row stride (ushorts) of transposed LDS tile

typedef __attribute__((ext_vector_type(16))) __bf16 v16bf;
typedef __attribute__((ext_vector_type(8)))  float  v8f;

union FragBF { v16bf v; __bf16 s[16]; };

__device__ inline v8f vzero8() { v8f z; for (int i = 0; i < 8; ++i) z[i] = 0.f; return z; }

__device__ inline v8f wmma_bf16(const FragBF& a, const FragBF& b, v8f c) {
  // 8 args: (neg_a, A, neg_b, B, c_mod, C, reuse_a, reuse_b)
  return __builtin_amdgcn_wmma_f32_16x16x32_bf16(false, a.v, false, b.v,
                                                 (short)0, c, false, false);
}

// A-matrix (16x32 bf16) fragment gather, contiguous k order used here (k = 16*grp + s):
// lanes 0-15: M=lane, V0..7 -> K {0..15}
// lanes 16-31: M=lane-16, V0..7 -> K {16..31}
// Per lane this is two contiguous 8-element runs -> vectorizes to ds_load_b128.
__device__ inline void load_a(FragBF& a, const __bf16* base,
                              int m, int stride, int k0, int grp) {
  const __bf16* p = base + m * stride + k0 + (grp ? 16 : 0);
#pragma unroll
  for (int i = 0; i < 8; ++i) a.s[i] = p[i];          // K run 0..7 (+grp*16)
#pragma unroll
  for (int i = 0; i < 8; ++i) a.s[8 + i] = p[8 + i];  // K run 8..15 (+grp*16)
}

// ---------------------------------------------------------------------------
// Kernel 1: per-batch prep.  q, gates=sigmoid(Wg qx+bg), R=bf16[16x512]
// ---------------------------------------------------------------------------
__global__ void prep_kernel(const float* __restrict__ x,
                            const float* __restrict__ Wq,
                            const float* __restrict__ Wkv,
                            const float* __restrict__ Wg,
                            const float* __restrict__ bg,
                            const int*   __restrict__ posp,
                            float* __restrict__ gates,
                            __bf16* __restrict__ Rbf) {
  const int b = blockIdx.x, t = threadIdx.x;          // 256 threads
  __shared__ float qx[DIM];
  __shared__ float qv[DIM];
  const int pos = posp[0];
  const float* xr = x + ((size_t)b * NCTX + pos) * DIM;
  qx[t] = xr[t];  qx[t + 256] = xr[t + 256];
  __syncthreads();

  for (int e = t; e < DIM; e += 256) {
    const float* wr = Wq + (size_t)e * DIM;
    const float* gr = Wg + (size_t)e * DIM;
    float sq = 0.f, sg = 0.f;
    for (int c = 0; c < DIM; ++c) { float xc = qx[c]; sq += wr[c] * xc; sg += gr[c] * xc; }
    qv[e] = sq;
    gates[b * DIM + e] = 1.f / (1.f + __expf(-(sg + bg[e])));
  }
  __syncthreads();

  const float SCALE = 0.125f;                         // 64^-0.5
  for (int idx = t; idx < HEADS * DIM; idx += 256) {  // R[h][c]
    int h = idx >> 9, c = idx & (DIM - 1);
    const float* wk = Wkv + (size_t)(h * 64) * DIM + c;  // K-half of Wkv
    float s = 0.f;
    for (int d = 0; d < 64; ++d) s += wk[(size_t)d * DIM] * qv[h * 64 + d];
    Rbf[(size_t)b * 16 * DIM + idx] = (__bf16)(s * SCALE);
  }
  for (int idx = t; idx < HEADS * DIM; idx += 256)    // zero pad heads 8..15
    Rbf[(size_t)b * 16 * DIM + HEADS * DIM + idx] = (__bf16)0.f;
}

// ---------------------------------------------------------------------------
// Kernel 2: S[b,h,j] = R_b @ C_b^T + bias (masked); per-block head maxima.
// grid (NJB, BATCH), 256 threads (8 waves, 32 j's per wave).
// ---------------------------------------------------------------------------
__global__ void s_kernel(const float* __restrict__ ctx,
                         const float* __restrict__ bias,
                         const __bf16* __restrict__ Rbf,
                         const unsigned char* __restrict__ mask,
                         const unsigned char* __restrict__ cmask,
                         float* __restrict__ S,
                         float* __restrict__ bmax) {
  const int b = blockIdx.y, jb = blockIdx.x;
  const int t = threadIdx.x, wave = t >> 5, lane = t & 31;
  const int nlo = lane & 15, grp = lane >> 4;
  __shared__ __bf16 Rl[16 * DIM];
  __shared__ float wmaxs[8][8];

  for (int i = t; i < 16 * DIM / 2; i += 256)
    ((unsigned int*)Rl)[i] = ((const unsigned int*)(Rbf + (size_t)b * 16 * DIM))[i];
  __syncthreads();

  const int jw = jb * JBLK + wave * 32;
  const float* row0 = ctx + ((size_t)(b * NCTX + jw + nlo)) * DIM;       // j = jw+n
  const float* row1 = row0 + (size_t)16 * DIM;                           // j = jw+16+n
  v8f acc0 = vzero8(), acc1 = vzero8();

  for (int ks = 0; ks < 16; ++ks) {                   // K = 512 in steps of 32
    const int k0 = ks * 32;
    FragBF a;  load_a(a, Rl, nlo, DIM, k0, grp);
    // B layout: lanes0-15 hold K=0..15, lanes16-31 K=16..31 -> 16 consecutive c
    const int c0 = k0 + grp * 16;
    FragBF b0, b1;
    const float4* p0 = (const float4*)(row0 + c0);
    const float4* p1 = (const float4*)(row1 + c0);
#pragma unroll
    for (int q = 0; q < 4; ++q) {
      float4 f0 = p0[q], f1 = p1[q];
      b0.s[q*4+0] = (__bf16)f0.x; b0.s[q*4+1] = (__bf16)f0.y;
      b0.s[q*4+2] = (__bf16)f0.z; b0.s[q*4+3] = (__bf16)f0.w;
      b1.s[q*4+0] = (__bf16)f1.x; b1.s[q*4+1] = (__bf16)f1.y;
      b1.s[q*4+2] = (__bf16)f1.z; b1.s[q*4+3] = (__bf16)f1.w;
    }
    acc0 = wmma_bf16(a, b0, acc0);
    acc1 = wmma_bf16(a, b1, acc1);
  }

  // epilogue: bias + mask, store S, per-head max
  const bool mb = mask[b] != 0;
  const int j0l = jw + nlo, j1l = jw + 16 + nlo;
  const bool c0m = mb && (cmask[b * NCTX + j0l] != 0);
  const bool c1m = mb && (cmask[b * NCTX + j1l] != 0);
  float hm[8];
#pragma unroll
  for (int r = 0; r < 8; ++r) {
    float s0 = c0m ? acc0[r] + bias[((size_t)(b * HEADS + r)) * NCTX + j0l] : -FLT_MAX;
    float s1 = c1m ? acc1[r] + bias[((size_t)(b * HEADS + r)) * NCTX + j1l] : -FLT_MAX;
    if (grp == 0) {                                   // lanes 0-15 hold heads 0..7
      S[((size_t)(b * HEADS + r)) * NCTX + j0l] = s0;
      S[((size_t)(b * HEADS + r)) * NCTX + j1l] = s1;
      hm[r] = fmaxf(s0, s1);
    } else hm[r] = -FLT_MAX;
  }
#pragma unroll
  for (int r = 0; r < 8; ++r) {
    float v = hm[r];
    for (int off = 1; off < 16; off <<= 1) v = fmaxf(v, __shfl_xor(v, off, 32));
    hm[r] = v;
  }
  if (lane == 0)
    for (int r = 0; r < 8; ++r) wmaxs[wave][r] = hm[r];
  __syncthreads();
  if (t < 8) {
    float mm = -FLT_MAX;
    for (int w = 0; w < 8; ++w) mm = fmaxf(mm, wmaxs[w][t]);
    bmax[(b * NJB + jb) * HEADS + t] = mm;
  }
}

// ---------------------------------------------------------------------------
// Kernel 3: global max per (b,h)
// ---------------------------------------------------------------------------
__global__ void maxred_kernel(const float* __restrict__ bmax, float* __restrict__ m) {
  const int t = threadIdx.x;                          // 128 = 16*8
  const int b = t >> 3, h = t & 7;
  float mm = -FLT_MAX;
  for (int i = 0; i < NJB; ++i) mm = fmaxf(mm, bmax[(b * NJB + i) * HEADS + h]);
  m[b * HEADS + h] = mm;
}

// ---------------------------------------------------------------------------
// Kernel 4: p = exp(s-m); Y_part = P @ C per 256-row block (WMMA), l partials.
// grid (NJB, BATCH), 256 threads (8 waves, each owns 4 feature tiles of 16).
// Context tile staged TRANSPOSED in LDS (Clt[c][k], padded stride) so the
// B fragment is 16 contiguous bf16 per lane -> two ds_load_b128.
// ---------------------------------------------------------------------------
__global__ void pv_kernel(const float* __restrict__ ctx,
                          const float* __restrict__ S,
                          const float* __restrict__ m,
                          float* __restrict__ lpart,
                          float* __restrict__ Ypart) {
  const int b = blockIdx.y, jb = blockIdx.x;
  const int t = threadIdx.x, wave = t >> 5, lane = t & 31;
  const int nlo = lane & 15, grp = lane >> 4;
  __shared__ __bf16 Pl[16 * JBLK];                    // bf16 probs, heads padded
  __shared__ __bf16 Clt[DIM * CSTR];                  // transposed bf16 tile [c][k]
  __shared__ float lsum[8];
  if (t < 8) lsum[t] = 0.f;
  __syncthreads();

  const int j0 = jb * JBLK;
  {
    float lp[8];
#pragma unroll
    for (int h = 0; h < 8; ++h) {
      float s = S[((size_t)(b * HEADS + h)) * NCTX + j0 + t];
      float p = __expf(s - m[b * HEADS + h]);
      Pl[h * JBLK + t] = (__bf16)p;
      lp[h] = p;
    }
#pragma unroll
    for (int h = 0; h < 8; ++h) atomicAdd(&lsum[h], lp[h]);
  }
  for (int r = 8; r < 16; ++r) Pl[r * JBLK + t] = (__bf16)0.f;  // pad heads
  __syncthreads();
  if (t < 8) lpart[(b * NJB + jb) * HEADS + t] = lsum[t];

  v8f acc[4];
  for (int tt = 0; tt < 4; ++tt) acc[tt] = vzero8();

  for (int ks = 0; ks < JBLK / 32; ++ks) {            // K = 256 rows in steps of 32
    __syncthreads();                                  // protect Clt reuse
    // stage 32 rows x 512 c, fp32 -> bf16, transposed into Clt[c][k]
    const float4* src = (const float4*)(ctx + ((size_t)(b * NCTX) + j0 + ks * 32) * DIM);
    for (int i = t; i < 32 * DIM / 4; i += 256) {
      const int row = i >> 7;                         // 128 float4 per row
      const int c   = (i & 127) * 4;
      float4 f = src[i];
      Clt[(c + 0) * CSTR + row] = (__bf16)f.x;
      Clt[(c + 1) * CSTR + row] = (__bf16)f.y;
      Clt[(c + 2) * CSTR + row] = (__bf16)f.z;
      Clt[(c + 3) * CSTR + row] = (__bf16)f.w;
    }
    __syncthreads();

    FragBF a;  load_a(a, Pl, nlo, JBLK, ks * 32, grp);
#pragma unroll
    for (int tt = 0; tt < 4; ++tt) {
      const int c0 = (wave * 4 + tt) * 16;
      FragBF bb;                                      // B[k][n] = Clt[c0+n][k]
      const __bf16* p = Clt + (c0 + nlo) * CSTR + grp * 16;
#pragma unroll
      for (int i = 0; i < 16; ++i) bb.s[i] = p[i];    // contiguous -> ds_load_b128 x2
      acc[tt] = wmma_bf16(a, bb, acc[tt]);
    }
  }

  if (grp == 0) {                                     // rows 0..7 = heads
    float* yp = Ypart + ((size_t)(b * NJB + jb)) * HEADS * DIM;
    for (int tt = 0; tt < 4; ++tt) {
      const int c = (wave * 4 + tt) * 16 + nlo;
      for (int r = 0; r < 8; ++r) yp[(size_t)r * DIM + c] = acc[tt][r];
    }
  }
}

// ---------------------------------------------------------------------------
// Kernel 5: reduce Y/l; out = Wo(gate * Wv_h Y_h / l_h) + bo.  One block per b.
// ---------------------------------------------------------------------------
__global__ void out_kernel(const float* __restrict__ Wkv,
                           const float* __restrict__ Wo,
                           const float* __restrict__ bo,
                           const float* __restrict__ gates,
                           const float* __restrict__ lpart,
                           const float* __restrict__ Ypart,
                           float* __restrict__ out) {
  const int b = blockIdx.x, t = threadIdx.x;          // 256 threads
  __shared__ float Y[HEADS * DIM];
  __shared__ float L[HEADS];
  __shared__ float g[DIM];
  if (t < 8) {
    float s = 0.f;
    for (int i = 0; i < NJB; ++i) s += lpart[(b * NJB + i) * HEADS + t];
    L[t] = s;
  }
  for (int idx = t; idx < HEADS * DIM; idx += 256) {
    float s = 0.f;
    for (int i = 0; i < NJB; ++i) s += Ypart[((size_t)(b * NJB + i)) * HEADS * DIM + idx];
    Y[idx] = s;
  }
  __syncthreads();
  for (int e = t; e < DIM; e += 256) {                // V projection + gate
    const int h = e >> 6;
    const float* wv = Wkv + (size_t)(DIM + e) * DIM;  // V-half of Wkv
    const float* yh = Y + h * DIM;
    float s = 0.f;
    for (int c = 0; c < DIM; ++c) s += wv[c] * yh[c];
    g[e] = (s / L[h]) * gates[b * DIM + e];
  }
  __syncthreads();
  for (int d = t; d < DIM; d += 256) {                // output projection
    const float* wo = Wo + (size_t)d * DIM;
    float s = bo[d];
    for (int e = 0; e < DIM; ++e) s += wo[e] * g[e];
    out[b * DIM + d] = s;
  }
}

// ---------------------------------------------------------------------------
extern "C" void kernel_launch(void* const* d_in, const int* in_sizes, int n_in,
                              void* d_out, int out_size, void* d_ws, size_t ws_size,
                              hipStream_t stream) {
  (void)in_sizes; (void)n_in; (void)out_size; (void)ws_size;
  const float* x     = (const float*)d_in[0];
  const float* ctx   = (const float*)d_in[1];
  const float* bias  = (const float*)d_in[2];
  const float* Wq    = (const float*)d_in[3];
  const float* Wkv   = (const float*)d_in[4];
  const float* Wo    = (const float*)d_in[5];
  const float* bo    = (const float*)d_in[6];
  const float* Wg    = (const float*)d_in[7];
  const float* bg    = (const float*)d_in[8];
  const unsigned char* mask  = (const unsigned char*)d_in[9];
  const unsigned char* cmask = (const unsigned char*)d_in[10];
  const int*   pos   = (const int*)d_in[11];
  float* out = (float*)d_out;

  char* w = (char*)d_ws;
  float*   gates = (float*)(w + 0);                   // 16*512 f32
  __bf16*  Rbf   = (__bf16*)(w + 32768);              // 16*16*512 bf16
  float*   S     = (float*)(w + 294912);              // 16*8*4096 f32
  float*   bmax  = (float*)(w + 2392064);             // 256*8 f32
  float*   mglob = (float*)(w + 2400256);             // 16*8 f32
  float*   lpart = (float*)(w + 2400768);             // 256*8 f32
  float*   Ypart = (float*)(w + 2408960);             // 256*8*512 f32

  prep_kernel  <<<BATCH, 256, 0, stream>>>(x, Wq, Wkv, Wg, bg, pos, gates, Rbf);
  s_kernel     <<<dim3(NJB, BATCH), 256, 0, stream>>>(ctx, bias, Rbf, mask, cmask, S, bmax);
  maxred_kernel<<<1, 128, 0, stream>>>(bmax, mglob);
  pv_kernel    <<<dim3(NJB, BATCH), 256, 0, stream>>>(ctx, S, mglob, lpart, Ypart);
  out_kernel   <<<BATCH, 256, 0, stream>>>(Wkv, Wo, bo, gates, lpart, Ypart, out);
}
